// ParallelMPNNModel_78297253806619
// MI455X (gfx1250) — matrix operand order repaired and checked by reading, not compiled, not run on hardware
//
#include <hip/hip_runtime.h>

// ---------------------------------------------------------------------------
// MPNN forward for MI455X (gfx1250): bf16 WMMA (16x16x32), wave32.
// - All GEMMs computed transposed (D = Wt x X^T): A-frags (weights) and
//   B-frags (activations) are 2x b128 LDS reads; epilogues are float4/uint4.
// - Generic GEMM stages tiles with double-buffered global_load_async_to_lds_b128
//   (ASYNCcnt) so DMA of tile k+32 overlaps WMMA on tile k.
// - Native __bf16 conversions (hardware RNE cvt where available).
// ---------------------------------------------------------------------------

typedef unsigned short u16;
typedef unsigned int u32;
typedef unsigned long long u64;
typedef __attribute__((ext_vector_type(16))) __bf16 v16bf;
typedef __attribute__((ext_vector_type(8))) __bf16 v8bf;
typedef __attribute__((ext_vector_type(8))) float v8f;

// ---------------- scalar helpers ----------------
__device__ __forceinline__ u16 f2bf(float f) {
  union { __bf16 b; u16 u; } c; c.b = (__bf16)f; return c.u;
}
__device__ __forceinline__ float bf2f(u16 h) {
  union { u16 u; __bf16 b; } c; c.u = h; return (float)c.b;
}
__device__ __forceinline__ v8f vzero8() {
  v8f z;
#pragma unroll
  for (int i = 0; i < 8; ++i) z[i] = 0.f;
  return z;
}
__device__ __forceinline__ uint4 pack8bf(const float* v) {
  union { v8bf b; uint4 q; } u;
#pragma unroll
  for (int i = 0; i < 8; ++i) u.b[i] = (__bf16)v[i];
  return u.q;
}

// ---------------- WMMA fragment helpers (CDNA5 wave32 layouts) ----------------
// A 16x32 bf16 (row-major [m][k] in LDS): lane m = lane&15;
//   element e <- k = 16*half + e (contiguous, matches B)
__device__ __forceinline__ v16bf load_a_frag(const u16* rowbase, int half) {
  union { uint4 q[2]; v16bf v; } u;
  u.q[0] = *(const uint4*)(rowbase + half * 16);
  u.q[1] = *(const uint4*)(rowbase + half * 16 + 8);
  return u.v;
}
// B 32x16 bf16 from row-major activations [n][k]: lane n = lane&15,
//   element e <- k = 16*half + e : one contiguous 32B read.
__device__ __forceinline__ v16bf load_row16(const u16* rp) {
  union { uint4 q[2]; v16bf v; } u;
  u.q[0] = *(const uint4*)rp;
  u.q[1] = *(const uint4*)(rp + 8);
  return u.v;
}
__device__ __forceinline__ v8f wmma_bf16(v16bf a, v16bf b, v8f c) {
  return __builtin_amdgcn_wmma_f32_16x16x32_bf16(false, a, false, b, (short)0, c,
                                                 false, false);
}

// ---- synchronous tile staging (msgs kernel): [128 x 32k] -> LDS [128][40] ----
__device__ __forceinline__ void stage_tile(const u16* __restrict__ g, size_t row0,
                                           int K, int k0, u16* lds, int tid) {
  const int r = tid >> 1, c = (tid & 1) * 16;
  const uint4* s = (const uint4*)(g + (row0 + r) * (size_t)K + k0 + c);
  uint4* d = (uint4*)(lds + r * 40 + c);
  d[0] = s[0];
  d[1] = s[1];
}

// ---- async tile staging (gemm kernel): ASYNCcnt-tracked DMA to LDS ----
__device__ __forceinline__ void stage_tile_async(const u16* __restrict__ g,
                                                 size_t row0, int K, int k0,
                                                 u16* lds, int tid) {
  const int r = tid >> 1, c = (tid & 1) * 16;
  const u16* src = g + (row0 + r) * (size_t)K + k0 + c;
  u16* dst = lds + r * 40 + c;
  const u32 ldsoff = (u32)(u64)(uintptr_t)dst;     // low 32 bits = LDS offset
  const u64 ga = (u64)(uintptr_t)src;
  asm volatile("global_load_async_to_lds_b128 %0, %1, off"
               :: "v"(ldsoff), "v"(ga) : "memory");
  asm volatile("global_load_async_to_lds_b128 %0, %1, off offset:16"
               :: "v"(ldsoff), "v"(ga) : "memory");
}
__device__ __forceinline__ void async_wait() {
  asm volatile("s_wait_asynccnt 0" ::: "memory");
}

// ---------------- encoders / map builders ----------------
__global__ __launch_bounds__(128) void encode_nodes_kernel(
    const int* __restrict__ x, const float* __restrict__ atom_tab,
    u16* __restrict__ zbf) {
  const int node = blockIdx.x;   // 4096
  const int h = threadIdx.x;     // 128
  float s = 0.f;
#pragma unroll
  for (int c = 0; c < 9; ++c) s += atom_tab[((size_t)c * 64 + x[node * 9 + c]) * 128 + h];
  zbf[(size_t)node * 256 + h] = f2bf(s);
  zbf[(size_t)node * 256 + 128 + h] = 0;  // hidden = 0 (bf16 zero)
}

__global__ __launch_bounds__(128) void encode_edges_kernel(
    const int* __restrict__ ea, const float* __restrict__ bond_tab,
    u16* __restrict__ edgebf) {
  const int e = blockIdx.x;      // 65536
  const int h = threadIdx.x;     // 128
  float s = 0.f;
#pragma unroll
  for (int c = 0; c < 3; ++c) s += bond_tab[((size_t)c * 8 + ea[e * 3 + c]) * 128 + h];
  edgebf[(size_t)e * 128 + h] = f2bf(s);
}

__global__ __launch_bounds__(256) void fill_map_kernel(int* __restrict__ emap) {
  const int i = blockIdx.x * 256 + threadIdx.x;
  if (i < 64 * 64 * 64) emap[i] = -1;
}

__global__ __launch_bounds__(256) void scatter_map_kernel(
    const int* __restrict__ ei, const int* __restrict__ bidx, int* __restrict__ emap) {
  const int e = blockIdx.x * 256 + threadIdx.x;
  if (e >= 65536) return;
  const int s = ei[e], d = ei[65536 + e];
  const int g = bidx[s];
  const int ls = s - g * 64, ld = d - g * 64;
  emap[(g * 64 + ls) * 64 + ld] = e;
}

// Convert f32 weight [K,128] -> transposed bf16 [128,K] (coalesced writes)
__global__ __launch_bounds__(256) void convert_bf16_t_kernel(
    const float* __restrict__ src, u16* __restrict__ dst, int K) {
  const int i = blockIdx.x * 256 + threadIdx.x;
  if (i >= K * 128) return;
  const int n = i / K, k = i - n * K;
  dst[i] = f2bf(src[(size_t)k * 128 + n]);
}

// ---------------- generic transposed WMMA GEMM ----------------
// out[q][p] = X[q,:K] . W[:,p] + bias[p]  (+out)   for q in [m0, m0+128)
// A = Wt [128 p][K], B = X [M][K]. Double-buffered async LDS staging.
template <bool ACCUM, bool OUTF, bool OUTB>
__global__ __launch_bounds__(256) void gemm_t_kernel(
    const u16* __restrict__ X, const u16* __restrict__ Wt,
    const float* __restrict__ bias, float* __restrict__ outF,
    u16* __restrict__ outB, int ldB, int colOffB, int K) {
  __shared__ u16 Xlds[2][128 * 40];
  __shared__ u16 Wlds[2][128 * 40];
  const int tid = threadIdx.x;
  const int lane = tid & 31, wv = tid >> 5;
  const int nl = lane & 15, half = lane >> 4;
  const size_t m0 = (size_t)blockIdx.x * 128;
  const int p0 = wv * 16 + half * 8;

  float bP[8];
  if (bias) {
    const float4 q0 = *(const float4*)(bias + p0);
    const float4 q1 = *(const float4*)(bias + p0 + 4);
    bP[0] = q0.x; bP[1] = q0.y; bP[2] = q0.z; bP[3] = q0.w;
    bP[4] = q1.x; bP[5] = q1.y; bP[6] = q1.z; bP[7] = q1.w;
  } else {
#pragma unroll
    for (int r = 0; r < 8; ++r) bP[r] = 0.f;
  }

  v8f acc[8];
#pragma unroll
  for (int i = 0; i < 8; ++i) acc[i] = vzero8();

  // prologue: DMA first tiles
  stage_tile_async(X, m0, K, 0, Xlds[0], tid);
  stage_tile_async(Wt, 0, K, 0, Wlds[0], tid);
  async_wait();
  __syncthreads();

  int cur = 0;
  for (int k0 = 0; k0 < K; k0 += 32) {
    const int nxt = cur ^ 1;
    if (k0 + 32 < K) {  // DMA next tiles while computing current
      stage_tile_async(X, m0, K, k0 + 32, Xlds[nxt], tid);
      stage_tile_async(Wt, 0, K, k0 + 32, Wlds[nxt], tid);
    }
    const v16bf af = load_a_frag(Wlds[cur] + (wv * 16 + nl) * 40, half);
#pragma unroll
    for (int nt = 0; nt < 8; ++nt) {
      const v16bf bf = load_row16(Xlds[cur] + (nt * 16 + nl) * 40 + 16 * half);
      acc[nt] = wmma_bf16(af, bf, acc[nt]);
    }
    async_wait();
    __syncthreads();
    cur = nxt;
  }

#pragma unroll
  for (int nt = 0; nt < 8; ++nt) {
    const size_t q = m0 + nt * 16 + nl;
    float v[8];
#pragma unroll
    for (int r = 0; r < 8; ++r) v[r] = acc[nt][r] + bP[r];
    if (ACCUM) {
      const float4 a0 = *(const float4*)(outF + q * 128 + p0);
      const float4 a1 = *(const float4*)(outF + q * 128 + p0 + 4);
      v[0] += a0.x; v[1] += a0.y; v[2] += a0.z; v[3] += a0.w;
      v[4] += a1.x; v[5] += a1.y; v[6] += a1.z; v[7] += a1.w;
    }
    if (OUTF) {
      *(float4*)(outF + q * 128 + p0) = make_float4(v[0], v[1], v[2], v[3]);
      *(float4*)(outF + q * 128 + p0 + 4) = make_float4(v[4], v[5], v[6], v[7]);
    }
    if (OUTB) {
      *(uint4*)(outB + q * (size_t)ldB + colOffB + p0) = pack8bf(v);
    }
  }
}

// ---------------- fused message kernel ----------------
// Per block: (graph b, 16-wide j tile). For each i-chunk of 8 (128 rows of
// (ii,jj) pairs): build relu(m1[j]+m2[i]+cbias+meN[e]) -> WMMA^T mlp1 -> relu
// -> WMMA^T mlp2 -> in-register max over ii(=nt) and chunks. Lane owns fixed
// (jj = lane&15, 8 consecutive out channels): no cross-wave reduction needed.
__global__ __launch_bounds__(256) void msgs_kernel(
    const float* __restrict__ m1, const float* __restrict__ m2,
    const u16* __restrict__ meN, const int* __restrict__ emap,
    const float* __restrict__ me_b, const float* __restrict__ mg_b,
    const u16* __restrict__ wt1, const float* __restrict__ b1,
    const u16* __restrict__ wt2, const float* __restrict__ b2,
    u16* __restrict__ mmax) {
  __shared__ u16 Xlds[128 * 40];
  __shared__ u16 Wlds[128 * 40];
  __shared__ u16 Ylds[128 * 136];   // Y[q=row][p=channel], padded
  __shared__ float cb[128];
  __shared__ int elds[128];

  const int tid = threadIdx.x;
  const int lane = tid & 31, wv = tid >> 5;
  const int b = blockIdx.x >> 2;
  const int j0 = (blockIdx.x & 3) * 16;
  const int nl = lane & 15, half = lane >> 4;
  const int p0 = wv * 16 + half * 8;

  if (tid < 128) cb[tid] = me_b[tid] + mg_b[tid];

  float bY[8];
  {
    const float4 q0 = *(const float4*)(b1 + p0);
    const float4 q1 = *(const float4*)(b1 + p0 + 4);
    bY[0] = q0.x; bY[1] = q0.y; bY[2] = q0.z; bY[3] = q0.w;
    bY[4] = q1.x; bY[5] = q1.y; bY[6] = q1.z; bY[7] = q1.w;
  }
  float maxreg[8];
#pragma unroll
  for (int r = 0; r < 8; ++r) maxreg[r] = -1e30f;
  __syncthreads();

  for (int i0 = 0; i0 < 64; i0 += 8) {
    if (tid < 128) {
      const int ii = tid >> 4, jj = tid & 15;
      elds[tid] = emap[(b * 64 + i0 + ii) * 64 + j0 + jj];
    }
    __syncthreads();

    // ---- GEMM1^T: Y[q][p] = relu-pre X @ mlp1 ----
    v8f acc[8];
#pragma unroll
    for (int i = 0; i < 8; ++i) acc[i] = vzero8();
    for (int k0 = 0; k0 < 128; k0 += 32) {
      stage_tile(wt1, 0, 128, k0, Wlds, tid);
      {  // build pre-activation rows X[q][k0..k0+31]
        const int r = tid >> 1;
        const int ii = r >> 4, jj = r & 15;
        const int i = i0 + ii, j = j0 + jj;
        const int e = elds[r];
        const int c0 = (tid & 1) * 16;
        const float* pm1 = m1 + (size_t)(b * 64 + j) * 128 + k0 + c0;
        const float* pm2 = m2 + (size_t)(b * 64 + i) * 128 + k0 + c0;
        const float* pcb = cb + k0 + c0;
        float xv[16];
#pragma unroll
        for (int t4 = 0; t4 < 4; ++t4) {
          const float4 a4 = *(const float4*)(pm1 + 4 * t4);
          const float4 b4 = *(const float4*)(pm2 + 4 * t4);
          const float4 c4 = *(const float4*)(pcb + 4 * t4);
          xv[4 * t4 + 0] = a4.x + b4.x + c4.x;
          xv[4 * t4 + 1] = a4.y + b4.y + c4.y;
          xv[4 * t4 + 2] = a4.z + b4.z + c4.z;
          xv[4 * t4 + 3] = a4.w + b4.w + c4.w;
        }
        if (e >= 0) {
          const u16* pme = meN + (size_t)e * 128 + k0 + c0;
          union { uint4 q[2]; u16 h[16]; } um;
          um.q[0] = *(const uint4*)pme;
          um.q[1] = *(const uint4*)(pme + 8);
#pragma unroll
          for (int kk = 0; kk < 16; ++kk) xv[kk] += bf2f(um.h[kk]);
        }
        union { u16 h[16]; uint4 q[2]; } uo;
#pragma unroll
        for (int kk = 0; kk < 16; ++kk) uo.h[kk] = f2bf(fmaxf(xv[kk], 0.f));
        uint4* d = (uint4*)(Xlds + r * 40 + c0);
        d[0] = uo.q[0];
        d[1] = uo.q[1];
      }
      __syncthreads();
      const v16bf af = load_a_frag(Wlds + (wv * 16 + nl) * 40, half);
#pragma unroll
      for (int nt = 0; nt < 8; ++nt) {
        const v16bf bf = load_row16(Xlds + (nt * 16 + nl) * 40 + 16 * half);
        acc[nt] = wmma_bf16(af, bf, acc[nt]);
      }
      __syncthreads();
    }
    // Y[q][p0..p0+7] = relu(acc + b1), one uint4 store per q-tile
#pragma unroll
    for (int nt = 0; nt < 8; ++nt) {
      const int q = nt * 16 + nl;
      float yv[8];
#pragma unroll
      for (int r = 0; r < 8; ++r) yv[r] = fmaxf(acc[nt][r] + bY[r], 0.f);
      *(uint4*)(Ylds + (size_t)q * 136 + p0) = pack8bf(yv);
    }
    __syncthreads();

    // ---- GEMM2^T: D2[p][q] with q = (ii=nt)*16 + (jj=nl) ----
    v8f acc2[8];
#pragma unroll
    for (int i = 0; i < 8; ++i) acc2[i] = vzero8();
    for (int k0 = 0; k0 < 128; k0 += 32) {
      stage_tile(wt2, 0, 128, k0, Wlds, tid);
      __syncthreads();
      const v16bf af = load_a_frag(Wlds + (wv * 16 + nl) * 40, half);
#pragma unroll
      for (int nt = 0; nt < 8; ++nt) {
        const v16bf bf = load_row16(Ylds + (size_t)(nt * 16 + nl) * 136 + k0 + 16 * half);
        acc2[nt] = wmma_bf16(af, bf, acc2[nt]);
      }
      __syncthreads();
    }
    // max over senders: nt = ii within chunk; chunks cover all i
#pragma unroll
    for (int nt = 0; nt < 8; ++nt)
#pragma unroll
      for (int r = 0; r < 8; ++r) maxreg[r] = fmaxf(maxreg[r], acc2[nt][r]);
  }

  // mmax[b*64 + j0 + jj][p0..p0+7] = maxreg + b2 (bias is i-invariant)
  {
    float ov[8];
    const float4 q0 = *(const float4*)(b2 + p0);
    const float4 q1 = *(const float4*)(b2 + p0 + 4);
    ov[0] = maxreg[0] + q0.x; ov[1] = maxreg[1] + q0.y;
    ov[2] = maxreg[2] + q0.z; ov[3] = maxreg[3] + q0.w;
    ov[4] = maxreg[4] + q1.x; ov[5] = maxreg[5] + q1.y;
    ov[6] = maxreg[6] + q1.z; ov[7] = maxreg[7] + q1.w;
    *(uint4*)(mmax + (size_t)(b * 64 + j0 + nl) * 128 + p0) = pack8bf(ov);
  }
}

// ---------------- relu + layernorm -> cat (bf16) ----------------
__global__ __launch_bounds__(128) void relu_ln_kernel(
    const float* __restrict__ t1, const float* __restrict__ g,
    const float* __restrict__ bb, u16* __restrict__ catbf, int coloff) {
  __shared__ float sred[128];
  const int row = blockIdx.x, tid = threadIdx.x;
  const float v = fmaxf(t1[(size_t)row * 128 + tid], 0.f);
  sred[tid] = v;
  __syncthreads();
#pragma unroll
  for (int s = 64; s > 0; s >>= 1) {
    if (tid < s) sred[tid] += sred[tid + s];
    __syncthreads();
  }
  const float mu = sred[0] * (1.f / 128.f);
  __syncthreads();
  const float d = v - mu;
  sred[tid] = d * d;
  __syncthreads();
#pragma unroll
  for (int s = 64; s > 0; s >>= 1) {
    if (tid < s) sred[tid] += sred[tid + s];
    __syncthreads();
  }
  const float var = sred[0] * (1.f / 128.f);
  const float val = d * rsqrtf(var + 1e-5f) * g[tid] + bb[tid];
  catbf[(size_t)row * 256 + coloff + tid] = f2bf(val);
}

// ---------------- graph head: mean -> gp1 relu -> gp2 ----------------
__global__ __launch_bounds__(128) void head_kernel(
    const float* __restrict__ hidden, const float* __restrict__ gp1_w,
    const float* __restrict__ gp1_b, const float* __restrict__ gp2_w,
    const float* __restrict__ gp2_b, float* __restrict__ out) {
  __shared__ float emb[128];
  __shared__ float h1[128];
  const int b = blockIdx.x, tid = threadIdx.x;
  float s = 0.f;
  for (int n = 0; n < 64; ++n) s += hidden[((size_t)(b * 64 + n)) * 128 + tid];
  emb[tid] = s * (1.f / 64.f);
  __syncthreads();
  float a = gp1_b[tid];
  for (int k = 0; k < 128; ++k) a += emb[k] * gp1_w[k * 128 + tid];
  h1[tid] = fmaxf(a, 0.f);
  __syncthreads();
  if (tid < 12) {
    float o = gp2_b[tid];
    for (int k = 0; k < 128; ++k) o += h1[k] * gp2_w[k * 12 + tid];
    out[b * 12 + tid] = o;
  }
}

// ---------------------------------------------------------------------------
// Host launcher
// ---------------------------------------------------------------------------
extern "C" void kernel_launch(void* const* d_in, const int* in_sizes, int n_in,
                              void* d_out, int out_size, void* d_ws, size_t ws_size,
                              hipStream_t stream) {
  (void)in_sizes; (void)n_in; (void)out_size; (void)ws_size;
  // --- input leaf indexing (insertion-order pytree flatten of setup_inputs) ---
  const float* atom_tab = (const float*)d_in[0];
  const float* bond_tab = (const float*)d_in[1];
  auto MP = [&](int l, int blk, int leaf) -> const float* {
    return (const float*)d_in[2 + (l * 2 + blk) * 18 + leaf];
  };
  // leaves: 0 m1_w 1 m1_b 2 m2_w 3 m2_b 4 me_w 5 me_b 6 mg_w 7 mg_b
  //         8 mlp1_w 9 mlp1_b 10 mlp2_w 11 mlp2_b 12 o1_w 13 o1_b 14 o2_w 15 o2_b
  //         16 ln_g 17 ln_b
  const float* red_w[2] = {(const float*)d_in[74], (const float*)d_in[76]};
  const float* red_b[2] = {(const float*)d_in[75], (const float*)d_in[77]};
  const float* gp1_w = (const float*)d_in[78];
  const float* gp1_b = (const float*)d_in[79];
  const float* gp2_w = (const float*)d_in[80];
  const float* gp2_b = (const float*)d_in[81];
  const int* x = (const int*)d_in[82];
  const int* edge_index = (const int*)d_in[83];
  const int* edge_attr = (const int*)d_in[84];
  const int* batch_idx = (const int*)d_in[85];

  // --- workspace layout ---
  size_t off = 0;
  auto alloc = [&](size_t bytes) -> void* {
    void* p = (char*)d_ws + off;
    off = (off + bytes + 255) & ~(size_t)255;
    return p;
  };
  u16* zbf    = (u16*)alloc((size_t)4096 * 256 * 2);    // z = [node_fts, hidden] bf16
  u16* edgebf = (u16*)alloc((size_t)65536 * 128 * 2);   // edge features bf16
  u16* meN    = (u16*)alloc((size_t)65536 * 128 * 2);   // edge @ Wme bf16
  int* emap   = (int*)alloc((size_t)64 * 64 * 64 * 4);  // dense edge-id map
  float* m1b  = (float*)alloc((size_t)4096 * 128 * 4);
  float* m2b  = (float*)alloc((size_t)4096 * 128 * 4);
  u16* mmax   = (u16*)alloc((size_t)4096 * 128 * 2);    // msgs max bf16
  float* t1   = (float*)alloc((size_t)4096 * 128 * 4);  // o1+o2 accumulator
  u16* catbf  = (u16*)alloc((size_t)4096 * 256 * 2);    // [r,p] concat bf16
  float* hidden = (float*)alloc((size_t)4096 * 128 * 4);
  u16* warena = (u16*)alloc((size_t)720896 * 2);        // transposed bf16 weights

  // --- convert weights to transposed bf16 (mg weight dead: graph_fts == 0) ---
  u16* cur = warena;
  u16 *w_m1[4], *w_m2[4], *w_me[4], *w_mlp1[4], *w_mlp2[4], *w_o1[4], *w_o2[4];
  u16* w_red[2];
  auto convT = [&](const float* src, int K) -> u16* {  // src [K,128] -> dst [128,K]
    u16* dst = cur;
    const int n = K * 128;
    cur += n;
    convert_bf16_t_kernel<<<(n + 255) / 256, 256, 0, stream>>>(src, dst, K);
    return dst;
  };
  for (int l = 0; l < 2; ++l)
    for (int blk = 0; blk < 2; ++blk) {
      const int idx = l * 2 + blk;
      w_m1[idx]   = convT(MP(l, blk, 0), 256);
      w_m2[idx]   = convT(MP(l, blk, 2), 256);
      w_me[idx]   = convT(MP(l, blk, 4), 128);
      w_mlp1[idx] = convT(MP(l, blk, 8), 128);
      w_mlp2[idx] = convT(MP(l, blk, 10), 128);
      w_o1[idx]   = convT(MP(l, blk, 12), 256);
      w_o2[idx]   = convT(MP(l, blk, 14), 128);
    }
  w_red[0] = convT(red_w[0], 256);
  w_red[1] = convT(red_w[1], 256);

  // --- encoders + edge map ---
  encode_nodes_kernel<<<4096, 128, 0, stream>>>(x, atom_tab, zbf);
  encode_edges_kernel<<<65536, 128, 0, stream>>>(edge_attr, bond_tab, edgebf);
  fill_map_kernel<<<(64 * 64 * 64 + 255) / 256, 256, 0, stream>>>(emap);
  scatter_map_kernel<<<256, 256, 0, stream>>>(edge_index, batch_idx, emap);

  // --- layers ---
  for (int l = 0; l < 2; ++l) {
    for (int blk = 0; blk < 2; ++blk) {
      const int idx = l * 2 + blk;
      // m1 = z @ Wm1 + b ; m2 = z @ Wm2 + b        (K = 256)
      gemm_t_kernel<false, true, false><<<32, 256, 0, stream>>>(
          zbf, w_m1[idx], MP(l, blk, 1), m1b, (u16*)0, 0, 0, 256);
      gemm_t_kernel<false, true, false><<<32, 256, 0, stream>>>(
          zbf, w_m2[idx], MP(l, blk, 3), m2b, (u16*)0, 0, 0, 256);
      // meN = edge_fts @ Wme (bias folded into msgs cbias)   (M = 65536)
      gemm_t_kernel<false, false, true><<<512, 256, 0, stream>>>(
          edgebf, w_me[idx], (const float*)0, (float*)0, meN, 128, 0, 128);
      // fused dense messages + mlp1/mlp2 + max over senders
      msgs_kernel<<<64 * 4, 256, 0, stream>>>(
          m1b, m2b, meN, emap, MP(l, blk, 5), MP(l, blk, 7),
          w_mlp1[idx], MP(l, blk, 9), w_mlp2[idx], MP(l, blk, 11), mmax);
      // t1 = z @ o1 + b ; t1 += msgs_max @ o2 + b
      gemm_t_kernel<false, true, false><<<32, 256, 0, stream>>>(
          zbf, w_o1[idx], MP(l, blk, 13), t1, (u16*)0, 0, 0, 256);
      gemm_t_kernel<true, true, false><<<32, 256, 0, stream>>>(
          mmax, w_o2[idx], MP(l, blk, 15), t1, (u16*)0, 0, 0, 128);
      // relu + layernorm -> cat[:, blk*128 : blk*128+128]
      relu_ln_kernel<<<4096, 128, 0, stream>>>(t1, MP(l, blk, 16), MP(l, blk, 17),
                                               catbf, blk * 128);
    }
    // hidden = cat @ red + b (f32), and bf16 copy into z[:,128:256]
    gemm_t_kernel<false, true, true><<<32, 256, 0, stream>>>(
        catbf, w_red[l], red_b[l], hidden, zbf, 256, 128, 256);
  }

  // --- graph head ---
  head_kernel<<<64, 128, 0, stream>>>(hidden, gp1_w, gp1_b, gp2_w, gp2_b,
                                      (float*)d_out);
}
